// Attention_1374389534886
// MI455X (gfx1250) — matrix operand order repaired and checked by reading, not compiled, not run on hardware
//
#include <hip/hip_runtime.h>
#include <hip/hip_bf16.h>

// ---------------------------------------------------------------------------
// CDNA5 / gfx1250 attention block. All matmuls run on v_wmma_f32_16x16x32_bf16
// (wave32 WMMA, f32 accumulate). Softmax is flash-style per 16-row q tile.
// ---------------------------------------------------------------------------

typedef __bf16 bf16;
typedef __attribute__((ext_vector_type(16))) __bf16 v16bf;
typedef __attribute__((ext_vector_type(8)))  __bf16 v8bf;
typedef __attribute__((ext_vector_type(8)))  float  v8f;

#define B_      2
#define N_      2048
#define DIM_    1024
#define HEADS_  16
#define DH_     64
#define INNER_  1024
#define SCALE_  8.0f
#define NEGMAX_ (-3.0e38f)

__device__ inline v8f zero8() {
  v8f z;
#pragma unroll
  for (int i = 0; i < 8; ++i) z[i] = 0.0f;
  return z;
}

__device__ inline v8f wmma_bf16(v16bf a, v16bf b, v8f c) {
  // D = A(16x32 bf16) * B(32x16 bf16) + C(16x16 f32)
  return __builtin_amdgcn_wmma_f32_16x16x32_bf16(
      /*neg_a=*/false, a, /*neg_b=*/false, b,
      /*c_mod=*/(short)0, c, /*reuse_a=*/false, /*reuse_b=*/false);
}

// A fragment (16x32, M x K): lane t holds row m = t&15; khalf = t>>4.
// elements e<8  -> A[m][khalf*8 + e]
// elements e>=8 -> A[m][16 + khalf*8 + (e-8)]
__device__ inline v16bf load_fragA(const bf16* __restrict__ base, int khalf) {
  v8bf lo = *(const v8bf*)(base + khalf * 8);
  v8bf hi = *(const v8bf*)(base + 16 + khalf * 8);
  v16bf r;
#pragma unroll
  for (int i = 0; i < 8; ++i) { r[i] = lo[i]; r[8 + i] = hi[i]; }
  return r;
}

// B fragment (32x16, K x N) from a column stored contiguously (B^T row):
// lane t holds col n = t&15; khalf = t>>4;
// elements e<8 -> B[khalf*8 + e][n]; e>=8 -> B[16 + khalf*8 + (e-8)][n]
__device__ inline v16bf load_fragB(const bf16* __restrict__ col, int khalf) {
  v8bf lo = *(const v8bf*)(col + khalf * 8);
  v8bf hi = *(const v8bf*)(col + 16 + khalf * 8);
  v16bf r;
#pragma unroll
  for (int i = 0; i < 8; ++i) { r[i] = lo[i]; r[8 + i] = hi[i]; }
  return r;
}

__device__ inline float redmax16(float v) {   // across 16-lane half
  v = fmaxf(v, __shfl_xor(v, 1, 32));
  v = fmaxf(v, __shfl_xor(v, 2, 32));
  v = fmaxf(v, __shfl_xor(v, 4, 32));
  v = fmaxf(v, __shfl_xor(v, 8, 32));
  return v;
}
__device__ inline float redsum16(float v) {
  v += __shfl_xor(v, 1, 32);
  v += __shfl_xor(v, 2, 32);
  v += __shfl_xor(v, 4, 32);
  v += __shfl_xor(v, 8, 32);
  return v;
}
__device__ inline float redsum32(float v) {
  v += __shfl_xor(v, 1, 32);
  v += __shfl_xor(v, 2, 32);
  v += __shfl_xor(v, 4, 32);
  v += __shfl_xor(v, 8, 32);
  v += __shfl_xor(v, 16, 32);
  return v;
}

// ---------------------------------------------------------------------------
// Transposed f32 -> bf16 weight conversion: W[K][N] -> WT[N][K]
// ---------------------------------------------------------------------------
__global__ __launch_bounds__(256) void cvt_t_kernel(const float* __restrict__ W,
                                                    bf16* __restrict__ WT,
                                                    int K, int N) {
  size_t i = (size_t)blockIdx.x * 256 + threadIdx.x;
  if (i >= (size_t)K * N) return;
  int k = (int)(i / N);
  int n = (int)(i % N);
  WT[(size_t)n * K + k] = (bf16)W[i];
}

// ---------------------------------------------------------------------------
// LayerNorm -> bf16. One block per row of 1024.
// ---------------------------------------------------------------------------
__global__ __launch_bounds__(256) void ln_kernel(const float* __restrict__ x,
                                                 const float* __restrict__ gamma,
                                                 bf16* __restrict__ xn) {
  const int row = blockIdx.x;
  const int tid = threadIdx.x;
  const float* xr = x + (size_t)row * DIM_;
  float4 v = reinterpret_cast<const float4*>(xr)[tid];
  float s  = v.x + v.y + v.z + v.w;
  float ss = v.x * v.x + v.y * v.y + v.z * v.z + v.w * v.w;
#pragma unroll
  for (int m = 1; m <= 16; m <<= 1) {
    s  += __shfl_xor(s, m, 32);
    ss += __shfl_xor(ss, m, 32);
  }
  __shared__ float rs[8], rss[8];
  const int lane = tid & 31, wv = tid >> 5;
  if (lane == 0) { rs[wv] = s; rss[wv] = ss; }
  __syncthreads();
  s = 0.0f; ss = 0.0f;
#pragma unroll
  for (int i = 0; i < 8; ++i) { s += rs[i]; ss += rss[i]; }
  const float mu  = s * (1.0f / DIM_);
  const float var = ss * (1.0f / DIM_) - mu * mu;
  const float inv = rsqrtf(var + 1e-5f);
  float4 g = reinterpret_cast<const float4*>(gamma)[tid];
  bf16* o = xn + (size_t)row * DIM_ + tid * 4;
  o[0] = (bf16)((v.x - mu) * inv * g.x);
  o[1] = (bf16)((v.y - mu) * inv * g.y);
  o[2] = (bf16)((v.z - mu) * inv * g.z);
  o[3] = (bf16)((v.w - mu) * inv * g.w);
}

// ---------------------------------------------------------------------------
// Generic bf16 WMMA GEMM: C[M][N] = A[M][K] * B[K][N], BT = B^T (N x K).
// Block = 8 waves; wave -> 16 rows x 64 cols (4 C fragments, A reuse x4).
// grid = (N/64, M/128)
// ---------------------------------------------------------------------------
__global__ __launch_bounds__(256) void gemm_bf16_kernel(
    const bf16* __restrict__ A, const bf16* __restrict__ BT,
    float* __restrict__ C, int M, int N, int K) {
  const int lane  = threadIdx.x & 31;
  const int wv    = threadIdx.x >> 5;
  const int m0    = blockIdx.y * 128 + wv * 16;
  const int n0    = blockIdx.x * 64;
  const int col   = lane & 15;
  const int khalf = lane >> 4;

  v8f acc[4];
#pragma unroll
  for (int c = 0; c < 4; ++c) acc[c] = zero8();

  const bf16* arow  = A + (size_t)(m0 + col) * K;
  const bf16* bcol0 = BT + (size_t)(n0 + col) * K;

  for (int k0 = 0; k0 < K; k0 += 32) {
    v16bf a = load_fragA(arow + k0, khalf);
#pragma unroll
    for (int c = 0; c < 4; ++c) {
      v16bf b = load_fragB(bcol0 + (size_t)c * 16 * K + k0, khalf);
      acc[c] = wmma_bf16(a, b, acc[c]);
    }
  }
#pragma unroll
  for (int c = 0; c < 4; ++c) {
#pragma unroll
    for (int r = 0; r < 8; ++r) {
      int m = m0 + r + 8 * khalf;
      int n = n0 + c * 16 + col;
      C[(size_t)m * N + n] = acc[c][r];
    }
  }
}

// ---------------------------------------------------------------------------
// Split heads, l2-normalize q/k (+ per-dim scales), transpose v.
// One wave per (b, h, n) row of 64. 8 waves/block.
// qn, kn : [b*H + h][n][64] bf16 ; vT : [b*H + h][64][N] bf16
// ---------------------------------------------------------------------------
__global__ __launch_bounds__(256) void prep_kernel(
    const float* __restrict__ qf, const float* __restrict__ kvf,
    const float* __restrict__ qsc, const float* __restrict__ ksc,
    bf16* __restrict__ qn, bf16* __restrict__ kn, bf16* __restrict__ vT) {
  const int lane = threadIdx.x & 31;
  const int wv   = threadIdx.x >> 5;
  const int gw   = blockIdx.x * 8 + wv;        // b*H*N waves
  const int n    = gw & (N_ - 1);
  const int h    = (gw >> 11) & (HEADS_ - 1);
  const int b    = gw >> 15;

  const float* q = qf  + (size_t)(b * N_ + n) * INNER_ + h * DH_;
  const float* k = kvf + (size_t)(b * N_ + n) * (2 * INNER_) + h * DH_;
  const float* v = k + INNER_;

  float q0 = q[lane], q1 = q[lane + 32];
  float k0 = k[lane], k1 = k[lane + 32];
  float v0 = v[lane], v1 = v[lane + 32];

  float qs = redsum32(q0 * q0 + q1 * q1);
  float ks = redsum32(k0 * k0 + k1 * k1);
  float qi = 1.0f / fmaxf(sqrtf(qs), 1e-12f);
  float ki = 1.0f / fmaxf(sqrtf(ks), 1e-12f);

  const size_t bhn = ((size_t)(b * HEADS_ + h) * N_ + n) * DH_;
  qn[bhn + lane]      = (bf16)(q0 * qi * qsc[lane]);
  qn[bhn + lane + 32] = (bf16)(q1 * qi * qsc[lane + 32]);
  kn[bhn + lane]      = (bf16)(k0 * ki * ksc[lane]);
  kn[bhn + lane + 32] = (bf16)(k1 * ki * ksc[lane + 32]);

  const size_t vtb = ((size_t)(b * HEADS_ + h) * DH_) * (size_t)N_ + n;
  vT[vtb + (size_t)lane * N_]        = (bf16)v0;
  vT[vtb + (size_t)(lane + 32) * N_] = (bf16)v1;
}

// ---------------------------------------------------------------------------
// Flash attention: one wave owns 16 q rows of one (b,h); loops over key tiles
// of 32 with online softmax. sim & p@v via WMMA; P transits LDS (C->A layout).
// ---------------------------------------------------------------------------
__global__ __launch_bounds__(256) void attn_kernel(
    const bf16* __restrict__ qn, const bf16* __restrict__ kn,
    const bf16* __restrict__ vT, const float* __restrict__ bias,
    const unsigned char* __restrict__ mask, bf16* __restrict__ out) {
  __shared__ __align__(16) bf16 plds[8][16][32];

  const int lane  = threadIdx.x & 31;
  const int wv    = threadIdx.x >> 5;
  const int gw    = blockIdx.x * 8 + wv;       // b*H*(N/16) waves
  const int qt    = gw & 127;
  const int h     = (gw >> 7) & (HEADS_ - 1);
  const int b     = gw >> 11;
  const int q0    = qt * 16;
  const int col   = lane & 15;
  const int khalf = lane >> 4;

  const size_t bh = (size_t)(b * HEADS_ + h);

  // q A-fragments (loaded once per tile)
  const bf16* qrow = qn + (bh * N_ + q0 + col) * DH_;
  v16bf aq0 = load_fragA(qrow, khalf);
  v16bf aq1 = load_fragA(qrow + 32, khalf);

  float mrun[8], lrun[8];
  v8f o[4];
#pragma unroll
  for (int r = 0; r < 8; ++r) { mrun[r] = NEGMAX_; lrun[r] = 0.0f; }
#pragma unroll
  for (int c = 0; c < 4; ++c) o[c] = zero8();

  const float* biasrow0 = bias + ((size_t)h * N_ + q0) * N_;
  const bf16* vbase     = vT + bh * DH_ * (size_t)N_;
  const int jmax = q0 + 15;

  for (int j0 = 0; j0 <= jmax; j0 += 32) {
    // ---- S = (q @ k^T) for 16 rows x 32 key columns ----
    const bf16* krow0 = kn + (bh * N_ + j0 + col) * DH_;
    const bf16* krow1 = krow0 + 16 * DH_;
    v8f s0 = zero8(), s1 = zero8();
    s0 = wmma_bf16(aq0, load_fragB(krow0, khalf), s0);
    s0 = wmma_bf16(aq1, load_fragB(krow0 + 32, khalf), s0);
    s1 = wmma_bf16(aq0, load_fragB(krow1, khalf), s1);
    s1 = wmma_bf16(aq1, load_fragB(krow1 + 32, khalf), s1);

    // prefetch next j-tile's bias (global_prefetch_b8)
    __builtin_prefetch(biasrow0 + (size_t)(8 * khalf) * N_ + j0 + 32 + col, 0, 1);

    // ---- scale + bias + key-mask + causal mask ----
    const bool km0 = mask[b * N_ + j0 + col] != 0;
    const bool km1 = mask[b * N_ + j0 + 16 + col] != 0;
#pragma unroll
    for (int r = 0; r < 8; ++r) {
      const int m = r + 8 * khalf;
      const float* bp = biasrow0 + (size_t)m * N_ + j0 + col;
      float e0 = s0[r] * SCALE_ + bp[0];
      float e1 = s1[r] * SCALE_ + bp[16];
      if (!km0 || (j0 + col) > (q0 + m))      e0 = NEGMAX_;
      if (!km1 || (j0 + 16 + col) > (q0 + m)) e1 = NEGMAX_;
      s0[r] = e0; s1[r] = e1;
    }

    // ---- online softmax (rows live across the 16-lane half) ----
    float alpha[8];
#pragma unroll
    for (int r = 0; r < 8; ++r) {
      float rm   = redmax16(fmaxf(s0[r], s1[r]));
      float mnew = fmaxf(mrun[r], rm);
      alpha[r]   = __expf(mrun[r] - mnew);
      float p0 = __expf(s0[r] - mnew);
      float p1 = __expf(s1[r] - mnew);
      s0[r] = p0; s1[r] = p1;
      lrun[r] = lrun[r] * alpha[r] + redsum16(p0 + p1);
      mrun[r] = mnew;
    }
#pragma unroll
    for (int c = 0; c < 4; ++c)
#pragma unroll
      for (int r = 0; r < 8; ++r) o[c][r] *= alpha[r];

    // ---- P (C layout) -> LDS -> A fragment ----
#pragma unroll
    for (int r = 0; r < 8; ++r) {
      const int m = r + 8 * khalf;
      plds[wv][m][col]      = (bf16)s0[r];
      plds[wv][m][16 + col] = (bf16)s1[r];
    }
    // per-wave private tile; DS ops are in-order per wave, fence the compiler
    asm volatile("s_wait_dscnt 0" ::: "memory");
    v16bf ap;
    {
      const bf16* prow = &plds[wv][col][0];
      v8bf lo = *(const v8bf*)(prow + khalf * 8);
      v8bf hi = *(const v8bf*)(prow + 16 + khalf * 8);
#pragma unroll
      for (int i = 0; i < 8; ++i) { ap[i] = lo[i]; ap[8 + i] = hi[i]; }
    }

    // ---- O += P @ V (4 d-column chunks of 16) ----
#pragma unroll
    for (int c = 0; c < 4; ++c) {
      const bf16* vr = vbase + (size_t)(c * 16 + col) * N_ + j0;
      o[c] = wmma_bf16(ap, load_fragB(vr, khalf), o[c]);
    }
  }

  // ---- finalize: O / l, write [b][n][h*64 + d] ----
  float inv[8];
#pragma unroll
  for (int r = 0; r < 8; ++r) inv[r] = (lrun[r] > 0.0f) ? (1.0f / lrun[r]) : 0.0f;
#pragma unroll
  for (int c = 0; c < 4; ++c)
#pragma unroll
    for (int r = 0; r < 8; ++r) {
      const int m = r + 8 * khalf;
      out[((size_t)b * N_ + q0 + m) * INNER_ + h * DH_ + c * 16 + col] =
          (bf16)(o[c][r] * inv[r]);
    }
}

// ---------------------------------------------------------------------------
// Launch pipeline
// ---------------------------------------------------------------------------
extern "C" void kernel_launch(void* const* d_in, const int* in_sizes, int n_in,
                              void* d_out, int out_size, void* d_ws, size_t ws_size,
                              hipStream_t stream) {
  const float* x     = (const float*)d_in[0];
  const float* gamma = (const float*)d_in[1];
  const float* Wq    = (const float*)d_in[2];
  const float* Wkv   = (const float*)d_in[3];
  const float* qsc   = (const float*)d_in[4];
  const float* ksc   = (const float*)d_in[5];
  const float* Wo    = (const float*)d_in[6];
  const float* bias  = (const float*)d_in[7];
  const unsigned char* mask = (const unsigned char*)d_in[8];
  float* out = (float*)d_out;

  const size_t BN = (size_t)B_ * N_;           // 4096 rows
  char* ws = (char*)d_ws;
  size_t off = 0;
  bf16* XN   = (bf16*)(ws + off);  off += BN * DIM_ * 2;                 // 8 MB
  bf16* WQT  = (bf16*)(ws + off);  off += (size_t)INNER_ * DIM_ * 2;     // 2 MB
  bf16* WKVT = (bf16*)(ws + off);  off += (size_t)2 * INNER_ * DIM_ * 2; // 4 MB
  bf16* WOT  = (bf16*)(ws + off);  off += (size_t)DIM_ * INNER_ * 2;     // 2 MB
  float* QF  = (float*)(ws + off); off += BN * INNER_ * 4;               // 16 MB
  float* KVF = (float*)(ws + off); off += BN * 2 * INNER_ * 4;           // 32 MB
  bf16* QN   = (bf16*)(ws + off);  off += BN * INNER_ * 2;               // 8 MB
  bf16* KN   = (bf16*)(ws + off);  off += BN * INNER_ * 2;               // 8 MB
  bf16* VT   = (bf16*)(ws + off);  off += BN * INNER_ * 2;               // 8 MB
  bf16* AO   = (bf16*)(ws + off);  off += BN * INNER_ * 2;               // 8 MB
  (void)ws_size; (void)in_sizes; (void)n_in; (void)out_size;

  // weights -> transposed bf16
  cvt_t_kernel<<<(DIM_ * INNER_ + 255) / 256, 256, 0, stream>>>(Wq, WQT, DIM_, INNER_);
  cvt_t_kernel<<<(DIM_ * 2 * INNER_ + 255) / 256, 256, 0, stream>>>(Wkv, WKVT, DIM_, 2 * INNER_);
  cvt_t_kernel<<<(INNER_ * DIM_ + 255) / 256, 256, 0, stream>>>(Wo, WOT, INNER_, DIM_);

  // layernorm
  ln_kernel<<<(int)BN, 256, 0, stream>>>(x, gamma, XN);

  // q / kv projections (WMMA GEMMs)
  {
    dim3 g(INNER_ / 64, (int)BN / 128);
    gemm_bf16_kernel<<<g, 256, 0, stream>>>(XN, WQT, QF, (int)BN, INNER_, DIM_);
  }
  {
    dim3 g(2 * INNER_ / 64, (int)BN / 128);
    gemm_bf16_kernel<<<g, 256, 0, stream>>>(XN, WKVT, KVF, (int)BN, 2 * INNER_, DIM_);
  }

  // l2norm + head split + V transpose
  prep_kernel<<<(int)(B_ * HEADS_ * N_ / 8), 256, 0, stream>>>(QF, KVF, qsc, ksc, QN, KN, VT);

  // flash attention
  attn_kernel<<<(int)(B_ * HEADS_ * (N_ / 16) / 8), 256, 0, stream>>>(QN, KN, VT, bias, mask, AO);

  // output projection -> f32 d_out
  {
    dim3 g(DIM_ / 64, (int)BN / 128);
    gemm_bf16_kernel<<<g, 256, 0, stream>>>(AO, WOT, out, (int)BN, DIM_, INNER_);
  }
}
